// LstmGeneticModel_30863634989519
// MI455X (gfx1250) — compile-verified, matrix operand order checked by reading, not run on hardware
//
#include <hip/hip_runtime.h>
#include <hip/hip_bf16.h>
#include <stdint.h>

#define Bsz   256
#define Tsz   4
#define DIN   3456
#define HENC  1024
#define HDEC  288
#define NLAY  8

typedef __attribute__((ext_vector_type(16))) __bf16 v16bf;
typedef __attribute__((ext_vector_type(8)))  float  v8f;

struct alignas(16) U4 { uint32_t a, b, c, d; };
struct alignas(16) F4 { float a, b, c, d; };

union AFrag {
    v16bf    v;
    uint16_t u[16];
};

struct BRaw { F4 x0, x1, x2, x3; };

static __device__ __forceinline__ BRaw load_braw(const float* p) {
    BRaw r;
    r.x0 = *(const F4*)(p);
    r.x1 = *(const F4*)(p + 4);
    r.x2 = *(const F4*)(p + 16);
    r.x3 = *(const F4*)(p + 20);
    return r;
}

static __device__ __forceinline__ v16bf cvt_braw(BRaw r) {
    v16bf b;
    b[0]  = (__bf16)r.x0.a; b[1]  = (__bf16)r.x0.b;
    b[2]  = (__bf16)r.x0.c; b[3]  = (__bf16)r.x0.d;
    b[4]  = (__bf16)r.x1.a; b[5]  = (__bf16)r.x1.b;
    b[6]  = (__bf16)r.x1.c; b[7]  = (__bf16)r.x1.d;
    b[8]  = (__bf16)r.x2.a; b[9]  = (__bf16)r.x2.b;
    b[10] = (__bf16)r.x2.c; b[11] = (__bf16)r.x2.d;
    b[12] = (__bf16)r.x3.a; b[13] = (__bf16)r.x3.b;
    b[14] = (__bf16)r.x3.c; b[15] = (__bf16)r.x3.d;
    return b;
}

// ---------------------------------------------------------------------------
// C(M,N) = A(M,K) * W(N,K)^T  [+ bias(N)  |  += C]     A: bf16, W: fp32, C: fp32
// Block: 256 threads = 8 waves; block tile 64(M) x 128(N); wave tile 16 x 64.
// Software-pipelined k-loop, phases fenced with sched_barriers:
//   [issue 18 loads for k+1] | [convert all 4 B frags of k] | [4 back-to-back
//   WMMAs] | [rotate].  Distinct live B fragments avoid the WMMA->VALU WAR
//   hazard NOP padding; load waits land one iteration later as graded waits.
// ---------------------------------------------------------------------------
__global__ __launch_bounds__(256)
void gemm_bf16_wmma(const __bf16* __restrict__ A, int lda,
                    const float* __restrict__ W, int ldw,
                    const float* __restrict__ bias,
                    float* __restrict__ C, int ldc,
                    int M, int N, int K, int accumulate)
{
    const int lane  = threadIdx.x & 31;
    const int wave  = threadIdx.x >> 5;
    const int wm    = wave & 3;
    const int wn    = wave >> 2;
    const int mBase = blockIdx.x * 64 + wm * 16;
    const int nBase = blockIdx.y * 128 + wn * 64;
    const int half  = lane >> 4;     // K-half selector per ISA 7.12.2 layout
    const int l15   = lane & 15;
    const int kOff  = half * 8;

    // Clamped coordinates -> every load in the k-loop is unconditional & in-bounds.
    const int aRow = min(mBase + l15, M - 1);
    const int nc0 = min(nBase + l15,      N - 1);
    const int nc1 = min(nBase + 16 + l15, N - 1);
    const int nc2 = min(nBase + 32 + l15, N - 1);
    const int nc3 = min(nBase + 48 + l15, N - 1);

    const __bf16* aPtr = A + (size_t)aRow * lda + kOff;
    const float*  wp0  = W + (size_t)nc0 * ldw + kOff;
    const float*  wp1  = W + (size_t)nc1 * ldw + kOff;
    const float*  wp2  = W + (size_t)nc2 * ldw + kOff;
    const float*  wp3  = W + (size_t)nc3 * ldw + kOff;

    v8f acc0, acc1, acc2, acc3;
    if (accumulate) {
        const int mRow0 = min(mBase + half * 8, M - 8);
#pragma unroll
        for (int r = 0; r < 8; ++r) {
            const size_t mo = (size_t)(mRow0 + r) * ldc;
            acc0[r] = C[mo + nc0];
            acc1[r] = C[mo + nc1];
            acc2[r] = C[mo + nc2];
            acc3[r] = C[mo + nc3];
        }
    } else {
        const float b0 = bias[nc0], b1 = bias[nc1], b2 = bias[nc2], b3 = bias[nc3];
#pragma unroll
        for (int r = 0; r < 8; ++r) { acc0[r] = b0; acc1[r] = b1; acc2[r] = b2; acc3[r] = b3; }
    }

    // pipeline prologue (K >= 32 for every call site)
    AFrag aCur;
    *(U4*)&aCur.u[0] = *(const U4*)(aPtr);
    *(U4*)&aCur.u[8] = *(const U4*)(aPtr + 16);
    BRaw r0 = load_braw(wp0);
    BRaw r1 = load_braw(wp1);
    BRaw r2 = load_braw(wp2);
    BRaw r3 = load_braw(wp3);

    for (int k = 0; k < K; k += 32) {
        // advance (clamped, scalar-uniform): last iteration harmlessly re-reads
        const int adv = (k + 32 < K) ? 32 : 0;
        aPtr += adv; wp0 += adv; wp1 += adv; wp2 += adv; wp3 += adv;

        // ---- load section: issue next iteration's 18 loads ----
        AFrag aNxt;
        *(U4*)&aNxt.u[0] = *(const U4*)(aPtr);
        *(U4*)&aNxt.u[8] = *(const U4*)(aPtr + 16);
        BRaw s0 = load_braw(wp0);
        BRaw s1 = load_braw(wp1);
        BRaw s2 = load_braw(wp2);
        BRaw s3 = load_braw(wp3);

        __builtin_amdgcn_sched_barrier(0);   // loads may not sink past here

        // ---- convert section: all 4 B fragments live in distinct registers ----
        v16bf bb0 = cvt_braw(r0);
        v16bf bb1 = cvt_braw(r1);
        v16bf bb2 = cvt_braw(r2);
        v16bf bb3 = cvt_braw(r3);

        __builtin_amdgcn_sched_barrier(0);   // keep converts out of the WMMA block

        // ---- WMMA section: 4 back-to-back, no intra-iteration hazards ----
        acc0 = __builtin_amdgcn_wmma_f32_16x16x32_bf16(
            false, aCur.v, false, bb0, (short)0, acc0, false, false);
        acc1 = __builtin_amdgcn_wmma_f32_16x16x32_bf16(
            false, aCur.v, false, bb1, (short)0, acc1, false, false);
        acc2 = __builtin_amdgcn_wmma_f32_16x16x32_bf16(
            false, aCur.v, false, bb2, (short)0, acc2, false, false);
        acc3 = __builtin_amdgcn_wmma_f32_16x16x32_bf16(
            false, aCur.v, false, bb3, (short)0, acc3, false, false);

        __builtin_amdgcn_sched_barrier(0);   // keep rotate below the WMMAs

        // ---- rotate pipeline (VALU moves double as WMMA WAR co-exec slots) ----
        aCur = aNxt;
        r0 = s0; r1 = s1; r2 = s2; r3 = s3;
    }

    // guarded store (only place correctness needs the real coordinates)
    const int n0 = nBase + l15;
#pragma unroll
    for (int r = 0; r < 8; ++r) {
        const int m = mBase + half * 8 + r;
        if (m < M) {
            const size_t mo = (size_t)m * ldc;
            if (n0      < N) C[mo + n0]      = acc0[r];
            if (n0 + 16 < N) C[mo + n0 + 16] = acc1[r];
            if (n0 + 32 < N) C[mo + n0 + 32] = acc2[r];
            if (n0 + 48 < N) C[mo + n0 + 48] = acc3[r];
        }
    }
}

// ---------------------------------------------------------------------------
// Fused LSTM cell: gates G(B,4H) -> c update + h output (bf16, (T,B,H) slice)
// ---------------------------------------------------------------------------
__global__ void lstm_cell(const float* __restrict__ G, float* __restrict__ Cst,
                          __bf16* __restrict__ Yt, int Bn, int H, int firstStep)
{
    const int idx = blockIdx.x * blockDim.x + threadIdx.x;
    if (idx >= Bn * H) return;
    const int b  = idx / H;
    const int hh = idx - b * H;
    const float* g = G + (size_t)b * 4 * H;
    const float gi = g[hh];
    const float gf = g[H + hh];
    const float gg = g[2 * H + hh];
    const float go = g[3 * H + hh];
    const float cprev = firstStep ? 0.0f : Cst[idx];
    const float si = 1.0f / (1.0f + __expf(-gi));
    const float sf = 1.0f / (1.0f + __expf(-gf));
    const float so = 1.0f / (1.0f + __expf(-go));
    const float c  = sf * cprev + si * tanhf(gg);
    const float h  = so * tanhf(c);
    Cst[idx] = c;
    Yt[idx]  = (__bf16)h;
}

// x (B,T,D) f32  ->  xbf (T,B,D) bf16
__global__ void convert_x(const float* __restrict__ x, __bf16* __restrict__ xbf,
                          int Bn, int Tn, int D)
{
    const int idx = blockIdx.x * blockDim.x + threadIdx.x;
    if (idx >= Bn * Tn * D) return;
    const int d  = idx % D;
    const int bt = idx / D;
    const int t  = bt % Tn;
    const int b  = bt / Tn;
    xbf[((size_t)t * Bn + b) * D + d] = (__bf16)x[idx];
}

// Y (T,B,H) bf16 -> Ah (B, T*H) bf16  (flatten order matches reference reshape)
__global__ void rearrange_head(const __bf16* __restrict__ Y, __bf16* __restrict__ Ah,
                               int Bn, int Tn, int H)
{
    const int idx = blockIdx.x * blockDim.x + threadIdx.x;
    if (idx >= Bn * Tn * H) return;
    const int h  = idx % H;
    const int bt = idx / H;
    const int t  = bt % Tn;
    const int b  = bt / Tn;
    Ah[idx] = Y[((size_t)t * Bn + b) * H + h];
}

__global__ void head_final(const float* __restrict__ H1, const float* __restrict__ W2,
                           const float* __restrict__ b2, float* __restrict__ out, int Bn)
{
    const int b = blockIdx.x * blockDim.x + threadIdx.x;
    if (b >= Bn) return;
    float acc = b2[0];
#pragma unroll
    for (int j = 0; j < 64; ++j) acc += H1[b * 64 + j] * W2[j];
    out[b] = acc;
}

// ---------------------------------------------------------------------------
static inline void run_gemm(const __bf16* A, int lda, const float* W, int ldw,
                            const float* bias, float* C, int ldc,
                            int M, int N, int K, int acc, hipStream_t s)
{
    dim3 grid((M + 63) / 64, (N + 127) / 128);
    gemm_bf16_wmma<<<grid, 256, 0, s>>>(A, lda, W, ldw, bias, C, ldc, M, N, K, acc);
}

extern "C" void kernel_launch(void* const* d_in, const int* in_sizes, int n_in,
                              void* d_out, int out_size, void* d_ws, size_t ws_size,
                              hipStream_t stream)
{
    const float* x        = (const float*)d_in[0];
    const float* eWih0    = (const float*)d_in[1];
    const float* eWhh0    = (const float*)d_in[2];
    const float* eb0      = (const float*)d_in[3];
    const float* eWih     = (const float*)d_in[4];
    const float* eWhh     = (const float*)d_in[5];
    const float* eb       = (const float*)d_in[6];
    const float* dWih0    = (const float*)d_in[7];
    const float* dWhh0    = (const float*)d_in[8];
    const float* db0      = (const float*)d_in[9];
    const float* dWih     = (const float*)d_in[10];
    const float* dWhh     = (const float*)d_in[11];
    const float* db       = (const float*)d_in[12];
    const float* W1       = (const float*)d_in[13];
    const float* b1       = (const float*)d_in[14];
    const float* W2       = (const float*)d_in[15];
    const float* b2       = (const float*)d_in[16];

    // workspace carve-up (256-B aligned)
    char*  ws  = (char*)d_ws;
    size_t off = 0;
    auto carve = [&](size_t bytes) -> void* {
        void* p = ws + off;
        off = (off + bytes + 255) & ~(size_t)255;
        return p;
    };
    __bf16* xbf = (__bf16*)carve((size_t)Tsz * Bsz * DIN  * 2);
    __bf16* Y0  = (__bf16*)carve((size_t)Tsz * Bsz * HENC * 2);
    __bf16* Y1  = (__bf16*)carve((size_t)Tsz * Bsz * HENC * 2);
    float*  Z   = (float*) carve((size_t)Tsz * Bsz * 4 * HENC * 4);
    float*  Cst = (float*) carve((size_t)Bsz * HENC * 4);
    __bf16* Ah  = (__bf16*)carve((size_t)Bsz * Tsz * HDEC * 2);
    float*  H1  = (float*) carve((size_t)Bsz * 64 * 4);

    {   // x -> (T,B,D) bf16
        const int n = Bsz * Tsz * DIN;
        convert_x<<<(n + 255) / 256, 256, 0, stream>>>(x, xbf, Bsz, Tsz, DIN);
    }

    // -------------------- encoder: 8 layers, H = 1024 --------------------
    const __bf16* in = xbf;
    int Kin = DIN;
    __bf16* out = Y0;
    for (int l = 0; l < NLAY; ++l) {
        const float* Wih = (l == 0) ? eWih0 : eWih + (size_t)(l - 1) * 4 * HENC * HENC;
        const float* Whh = (l == 0) ? eWhh0 : eWhh + (size_t)(l - 1) * 4 * HENC * HENC;
        const float* bb  = (l == 0) ? eb0   : eb   + (size_t)(l - 1) * 4 * HENC;
        // all-timestep input projection: Z(T*B, 4H) = X * Wih^T + b
        run_gemm(in, Kin, Wih, Kin, bb, Z, 4 * HENC, Tsz * Bsz, 4 * HENC, Kin, 0, stream);
        for (int t = 0; t < Tsz; ++t) {
            if (t > 0)   // Z_t += h_{t-1} * Whh^T
                run_gemm(out + (size_t)(t - 1) * Bsz * HENC, HENC, Whh, HENC, nullptr,
                         Z + (size_t)t * Bsz * 4 * HENC, 4 * HENC,
                         Bsz, 4 * HENC, HENC, 1, stream);
            const int n = Bsz * HENC;
            lstm_cell<<<(n + 255) / 256, 256, 0, stream>>>(
                Z + (size_t)t * Bsz * 4 * HENC, Cst,
                out + (size_t)t * Bsz * HENC, Bsz, HENC, t == 0);
        }
        in  = out;
        Kin = HENC;
        out = (out == Y0) ? Y1 : Y0;
    }

    // -------------------- decoder: 8 layers, H = 288 --------------------
    for (int l = 0; l < NLAY; ++l) {
        const float* Wih = (l == 0) ? dWih0 : dWih + (size_t)(l - 1) * 4 * HDEC * HDEC;
        const float* Whh = (l == 0) ? dWhh0 : dWhh + (size_t)(l - 1) * 4 * HDEC * HDEC;
        const float* bb  = (l == 0) ? db0   : db   + (size_t)(l - 1) * 4 * HDEC;
        run_gemm(in, Kin, Wih, Kin, bb, Z, 4 * HDEC, Tsz * Bsz, 4 * HDEC, Kin, 0, stream);
        for (int t = 0; t < Tsz; ++t) {
            if (t > 0)
                run_gemm(out + (size_t)(t - 1) * Bsz * HDEC, HDEC, Whh, HDEC, nullptr,
                         Z + (size_t)t * Bsz * 4 * HDEC, 4 * HDEC,
                         Bsz, 4 * HDEC, HDEC, 1, stream);
            const int n = Bsz * HDEC;
            lstm_cell<<<(n + 255) / 256, 256, 0, stream>>>(
                Z + (size_t)t * Bsz * 4 * HDEC, Cst,
                out + (size_t)t * Bsz * HDEC, Bsz, HDEC, t == 0);
        }
        in  = out;
        Kin = HDEC;
        out = (out == Y0) ? Y1 : Y0;
    }

    // -------------------- FC head --------------------
    {
        const int n = Bsz * Tsz * HDEC;
        rearrange_head<<<(n + 255) / 256, 256, 0, stream>>>(in, Ah, Bsz, Tsz, HDEC);
    }
    run_gemm(Ah, Tsz * HDEC, W1, Tsz * HDEC, b1, H1, 64, Bsz, 64, Tsz * HDEC, 0, stream);
    head_final<<<1, 256, 0, stream>>>(H1, W2, b2, (float*)d_out, Bsz);
}
